// ProductKeyRouter_34961033789979
// MI455X (gfx1250) — matrix operand order repaired and checked by reading, not compiled, not run on hardware
//
#include <hip/hip_runtime.h>

// Problem constants (match reference)
#define NTOK  16384
#define DM    2048
#define QH    512          // d_query per head
#define NH    4
#define HQ    2048         // NH*QH
#define SK    256          // sub-keys per table
#define CH    256          // Q/2, contraction dim of scoring
#define KC    8            // stage-1 top-k
#define TPK   4            // final top-k
#define BNEPS 1e-5f
#define KSTEPS (DM / 32)   // 64 k-tiles in the projection GEMM

typedef __attribute__((ext_vector_type(16))) __bf16 bf16x16;
typedef __attribute__((ext_vector_type(8)))  __bf16 bf16x8;
typedef __attribute__((ext_vector_type(8)))  float  f32x8;

// ---- WMMA fragment helpers (contiguous k order: k = 16*(lane/16) + s for A and B, wave32) ----------

// A matrix 16x32 bf16, source row-major [16][stride].
// lane<16: M=lane, K = 0..15; lane>=16: M=lane-16, K = 16..31
__device__ __forceinline__ bf16x16 fragA_16x32(const __bf16* tile, int stride, int lane) {
  const __bf16* row = tile + (size_t)(lane & 15) * stride;
  const int kb = (lane & 16) ? 16 : 0;
  bf16x8 lo = *(const bf16x8*)(row + kb);
  bf16x8 hi = *(const bf16x8*)(row + kb + 8);
  bf16x16 r;
#pragma unroll
  for (int i = 0; i < 8; ++i) { r[i] = lo[i]; r[i + 8] = hi[i]; }
  return r;
}

// B matrix 32x16 bf16, source stored N-major: [16 n][stride>=32 k].
// lane<16: N=lane, K=0..15 ; lane>=16: N=lane-16, K=16..31  (VGPR v = K{2v,2v+1})
__device__ __forceinline__ bf16x16 fragB_32x16(const __bf16* tile, int stride, int lane) {
  const __bf16* p = tile + (size_t)(lane & 15) * stride + ((lane & 16) ? 16 : 0);
  return *(const bf16x16*)p;
}

__device__ __forceinline__ f32x8 wmma_bf16(bf16x16 a, bf16x16 b, f32x8 c) {
  return __builtin_amdgcn_wmma_f32_16x16x32_bf16(false, a, false, b, (short)0, c,
                                                 false, false);
}

// ---- kernel 0: zero stats accumulators -------------------------------------
__global__ void zero_kernel(float* p, int n) {
  int i = blockIdx.x * 256 + threadIdx.x;
  if (i < n) p[i] = 0.0f;
}

// ---- kernel 0b: one-time W transpose+convert: WbfT[col][d] = bf16(Wq[h][d][q])
// 64x64 tiles through LDS (padded stride to dodge bank conflicts).
__global__ __launch_bounds__(256) void convert_w_kernel(const float* __restrict__ Wq,
                                                        __bf16* __restrict__ wbfT) {
  __shared__ __bf16 T[64][66];
  const int col0 = blockIdx.x * 64;     // output row block (col = h*QH + q)
  const int d0   = blockIdx.y * 64;
  const int h    = col0 >> 9;
  const int q0   = col0 & (QH - 1);
  const int t    = threadIdx.x;
  // load 64(d) x 64(q) floats, convert, store to LDS
  {
    const int dr = t >> 2;              // 0..63
    const int qc = (t & 3) * 16;        // 0..48
    const float* src = Wq + ((size_t)(h * DM + d0 + dr)) * QH + q0 + qc;
#pragma unroll
    for (int i = 0; i < 16; ++i) T[dr][qc + i] = (__bf16)src[i];
  }
  __syncthreads();
  // write transposed: WbfT[col0+cr][d0 + dd..dd+15]
  {
    const int cr = t >> 2;              // 0..63
    const int dd = (t & 3) * 16;
    bf16x8 p0, p1;
#pragma unroll
    for (int i = 0; i < 8; ++i) { p0[i] = T[dd + i][cr]; p1[i] = T[dd + 8 + i][cr]; }
    __bf16* dst = wbfT + (size_t)(col0 + cr) * DM + d0 + dd;
    *(bf16x8*)dst = p0; *(bf16x8*)(dst + 8) = p1;
  }
}

// ---- kernel 1: q = x @ Wq + bq  (bf16 WMMA, f32 accum) ---------------------
// Block tile 128(M) x 128(N), K-step 32, double-buffered LDS, register-staged
// software pipeline. 8 waves, each 32x64 (2x4 16x16 tiles).
// Epilogue: fused BN batch-statistics (global f32 atomics) + bf16 q store.
__global__ __launch_bounds__(256) void gemm_q_kernel(const float* __restrict__ x,
                                                     const __bf16* __restrict__ wbfT,
                                                     const float* __restrict__ bq,
                                                     __bf16* __restrict__ qbf,
                                                     float* __restrict__ sum,
                                                     float* __restrict__ ssq) {
  __shared__ __attribute__((aligned(32))) __bf16 As[2][128 * 32]; // [m][k]
  __shared__ __attribute__((aligned(32))) __bf16 Bs[2][128 * 32]; // [n][k]

  const int bm = blockIdx.x * 128;        // token base
  const int bn = blockIdx.y * 128;        // output-column base (= h*QH + q)
  const int tid = threadIdx.x;
  const int lane = tid & 31, wid = tid >> 5;
  const int wm = (wid & 3) * 32;          // wave m offset
  const int wn = (wid >> 2) * 64;         // wave n offset

  // per-thread staging coordinates
  const int ar = tid >> 1, ak = (tid & 1) * 16;   // A: 16 consecutive floats
  const int br = tid >> 1, bk = (tid & 1) * 16;   // B: 16 consecutive bf16
  const float*  aptr = x    + (size_t)(bm + ar) * DM + ak;
  const __bf16* bptr = wbfT + (size_t)(bn + br) * DM + bk;

  float  areg[16];
  bf16x8 breg0, breg1;

  // ---- prologue: load + stage k-tile 0 into buffer 0
#pragma unroll
  for (int i = 0; i < 16; ++i) areg[i] = aptr[i];
  breg0 = *(const bf16x8*)bptr;
  breg1 = *(const bf16x8*)(bptr + 8);
  {
    __bf16* ad = &As[0][ar * 32 + ak];
    bf16x8 p0, p1;
#pragma unroll
    for (int i = 0; i < 8; ++i) { p0[i] = (__bf16)areg[i]; p1[i] = (__bf16)areg[8 + i]; }
    *(bf16x8*)ad = p0; *(bf16x8*)(ad + 8) = p1;
    __bf16* bd = &Bs[0][br * 32 + bk];
    *(bf16x8*)bd = breg0; *(bf16x8*)(bd + 8) = breg1;
  }
  __syncthreads();

  f32x8 acc[2][4] = {};

  for (int kt = 0; kt < KSTEPS; ++kt) {
    const int cur = kt & 1;
    const bool more = (kt + 1) < KSTEPS;

    // issue next tile's global loads early (latency hidden behind WMMAs)
    if (more) {
      const float*  ap = aptr + (kt + 1) * 32;
      const __bf16* bp = bptr + (kt + 1) * 32;
#pragma unroll
      for (int i = 0; i < 16; ++i) areg[i] = ap[i];
      breg0 = *(const bf16x8*)bp;
      breg1 = *(const bf16x8*)(bp + 8);
    }
    if (kt + 2 < KSTEPS) {  // pull k+2 tiles toward the caches (global_prefetch_b8)
      __builtin_prefetch(aptr + (kt + 2) * 32, 0, 3);
      __builtin_prefetch(bptr + (kt + 2) * 32, 0, 3);
    }

    // compute on current buffer
#pragma unroll
    for (int mi = 0; mi < 2; ++mi) {
      bf16x16 a = fragA_16x32(&As[cur][(wm + mi * 16) * 32], 32, lane);
#pragma unroll
      for (int ni = 0; ni < 4; ++ni) {
        bf16x16 b = fragB_32x16(&Bs[cur][(wn + ni * 16) * 32], 32, lane);
        acc[mi][ni] = wmma_bf16(a, b, acc[mi][ni]);
      }
    }

    // stage next tile into the other buffer
    if (more) {
      const int nxt = cur ^ 1;
      __bf16* ad = &As[nxt][ar * 32 + ak];
      bf16x8 p0, p1;
#pragma unroll
      for (int i = 0; i < 8; ++i) { p0[i] = (__bf16)areg[i]; p1[i] = (__bf16)areg[8 + i]; }
      *(bf16x8*)ad = p0; *(bf16x8*)(ad + 8) = p1;
      __bf16* bd = &Bs[nxt][br * 32 + bk];
      *(bf16x8*)bd = breg0; *(bf16x8*)(bd + 8) = breg1;
    }
    __syncthreads();
  }

  // epilogue: + bq, store bf16 q, accumulate BN batch statistics
  const int roff = (lane & 16) ? 8 : 0;
#pragma unroll
  for (int ni = 0; ni < 4; ++ni) {
    const int colg = bn + wn + ni * 16 + (lane & 15);
    const float bias = bq[colg];            // bq flat [NH*QH] == column index
    float s = 0.0f, s2 = 0.0f;
#pragma unroll
    for (int mi = 0; mi < 2; ++mi) {
      const int rowg = bm + wm + mi * 16 + roff;
#pragma unroll
      for (int r = 0; r < 8; ++r) {
        const float v = acc[mi][ni][r] + bias;
        qbf[(size_t)(rowg + r) * HQ + colg] = (__bf16)v;
        s += v; s2 += v * v;
      }
    }
    atomicAdd(&sum[colg], s);
    atomicAdd(&ssq[colg], s2);
  }
}

// ---- kernel 3: fold BN affine into bf16 key tables + per-key bias ----------
__global__ __launch_bounds__(256) void fold_kernel(const float* __restrict__ sum,
                                                   const float* __restrict__ ssq,
                                                   const float* __restrict__ gamma,
                                                   const float* __restrict__ beta,
                                                   const float* __restrict__ K1,
                                                   const float* __restrict__ K2,
                                                   __bf16* __restrict__ k1f,
                                                   __bf16* __restrict__ k2f,
                                                   float* __restrict__ bias1,
                                                   float* __restrict__ bias2) {
  __shared__ float red1[256], red2[256];
  const int b = blockIdx.x;      // h*SK + k
  const int h = b >> 8;
  const int k = b & (SK - 1);
  const int c = threadIdx.x;

  {
    const int ch = h * QH + c;
    const float mu  = sum[ch] * (1.0f / NTOK);
    const float var = ssq[ch] * (1.0f / NTOK) - mu * mu;
    const float a   = rsqrtf(var + BNEPS) * gamma[ch];
    const float bb  = beta[ch] - mu * a;
    const float kv  = K1[k * CH + c];
    k1f[(size_t)b * CH + c] = (__bf16)(a * kv);
    red1[c] = bb * kv;
  }
  {
    const int ch = h * QH + CH + c;
    const float mu  = sum[ch] * (1.0f / NTOK);
    const float var = ssq[ch] * (1.0f / NTOK) - mu * mu;
    const float a   = rsqrtf(var + BNEPS) * gamma[ch];
    const float bb  = beta[ch] - mu * a;
    const float kv  = K2[k * CH + c];
    k2f[(size_t)b * CH + c] = (__bf16)(a * kv);
    red2[c] = bb * kv;
  }
  __syncthreads();
  for (int s = 128; s > 0; s >>= 1) {
    if (c < s) { red1[c] += red1[c + s]; red2[c] += red2[c + s]; }
    __syncthreads();
  }
  if (c == 0) { bias1[b] = red1[0]; bias2[b] = red2[0]; }
}

// ---- kernel 4: scoring WMMA + top-8/top-8 -> top-4 -> softmax --------------
// Block: 16 tokens x 1 head. 8 waves: wid>>2 = table, (wid&3)*64 = key base.
__global__ __launch_bounds__(256) void score_kernel(const __bf16* __restrict__ qbf,
                                                    const __bf16* __restrict__ k1f,
                                                    const __bf16* __restrict__ k2f,
                                                    const float* __restrict__ bias1,
                                                    const float* __restrict__ bias2,
                                                    float* __restrict__ out) {
  __shared__ __attribute__((aligned(32))) __bf16 qa[16 * 512]; // raw q tile, both halves
  __shared__ float sc[16 * 512];                               // s1 | s2 per token

  const int n0  = blockIdx.x * 16;
  const int h   = blockIdx.y;
  const int tid = threadIdx.x;
  const int lane = tid & 31, wid = tid >> 5;

  // stage q tile (already bf16; BN folded into keys): pure 64B copy per thread
  {
    const int r  = tid >> 4;            // 0..15
    const int c0 = (tid & 15) * 32;     // 0..480
    const __bf16* src = qbf + (size_t)(n0 + r) * HQ + h * QH + c0;
    __bf16* dst = qa + r * 512 + c0;
#pragma unroll
    for (int v = 0; v < 4; ++v)
      *(bf16x8*)(dst + v * 8) = *(const bf16x8*)(src + v * 8);
  }
  __syncthreads();

  const int table = wid >> 2;                 // 0 -> s1, 1 -> s2
  const int kb0   = (wid & 3) * 64;           // 64 keys per wave
  const __bf16* Kf   = table ? k2f : k1f;
  const float*  bias = table ? bias2 : bias1;

  f32x8 acc[4] = {};
  for (int k0 = 0; k0 < CH; k0 += 32) {
    bf16x16 a = fragA_16x32(qa + table * CH + k0, 512, lane);
#pragma unroll
    for (int ni = 0; ni < 4; ++ni) {
      const int key = kb0 + ni * 16 + (lane & 15);
      const __bf16* bp = Kf + ((size_t)(h * SK + key)) * CH + k0 + ((lane & 16) ? 16 : 0);
      bf16x16 b = *(const bf16x16*)bp;        // contiguous 32B from L2-resident table
      acc[ni] = wmma_bf16(a, b, acc[ni]);
    }
  }

  const int roff = (lane & 16) ? 8 : 0;
#pragma unroll
  for (int ni = 0; ni < 4; ++ni) {
    const int key = kb0 + ni * 16 + (lane & 15);
    const float bv = bias[h * SK + key];
#pragma unroll
    for (int r = 0; r < 8; ++r)
      sc[(roff + r) * 512 + table * CH + key] = acc[ni][r] + bv;
  }
  __syncthreads();

  // ---- per-token selection (one thread per token, register-resident) ----
  if (tid < 16) {
    const float* row = sc + tid * 512;
    float v1[KC], v2[KC];
    int   i1[KC], i2[KC];
#pragma unroll
    for (int j = 0; j < KC; ++j) { v1[j] = -1e30f; v2[j] = -1e30f; i1[j] = 0; i2[j] = 0; }

    for (int k = 0; k < SK; ++k) {
      float s = row[k];
      if (s > v1[KC - 1]) {
        v1[KC - 1] = s; i1[KC - 1] = k;
#pragma unroll
        for (int j = KC - 1; j > 0; --j)
          if (v1[j] > v1[j - 1]) {
            float tv = v1[j]; v1[j] = v1[j - 1]; v1[j - 1] = tv;
            int   ti = i1[j]; i1[j] = i1[j - 1]; i1[j - 1] = ti;
          }
      }
    }
    for (int k = 0; k < SK; ++k) {
      float s = row[CH + k];
      if (s > v2[KC - 1]) {
        v2[KC - 1] = s; i2[KC - 1] = k;
#pragma unroll
        for (int j = KC - 1; j > 0; --j)
          if (v2[j] > v2[j - 1]) {
            float tv = v2[j]; v2[j] = v2[j - 1]; v2[j - 1] = tv;
            int   ti = i2[j]; i2[j] = i2[j - 1]; i2[j - 1] = ti;
          }
      }
    }

    // 64 candidate combos -> top-4 (ascending combo index preserves tie order)
    float fv[TPK]; int fi[TPK];
#pragma unroll
    for (int j = 0; j < TPK; ++j) { fv[j] = -1e30f; fi[j] = 0; }
#pragma unroll
    for (int a_ = 0; a_ < KC; ++a_) {
#pragma unroll
      for (int b_ = 0; b_ < KC; ++b_) {
        float s = v1[a_] + v2[b_];
        if (s > fv[TPK - 1]) {
          fv[TPK - 1] = s; fi[TPK - 1] = a_ * KC + b_;
#pragma unroll
          for (int j = TPK - 1; j > 0; --j)
            if (fv[j] > fv[j - 1]) {
              float tv = fv[j]; fv[j] = fv[j - 1]; fv[j - 1] = tv;
              int   ti = fi[j]; fi[j] = fi[j - 1]; fi[j - 1] = ti;
            }
        }
      }
    }

    // softmax over the 4 (fv sorted descending -> fv[0] is max)
    float e[TPK], ssum = 0.0f;
#pragma unroll
    for (int j = 0; j < TPK; ++j) { e[j] = __expf(fv[j] - fv[0]); ssum += e[j]; }
    const float inv = 1.0f / ssum;

    const size_t obase = ((size_t)(n0 + tid) * NH + h) * TPK;
    const size_t eoff  = (size_t)NTOK * NH * TPK;
#pragma unroll
    for (int j = 0; j < TPK; ++j) {
      out[obase + j] = e[j] * inv;
      const int c1 = fi[j] >> 3, c2 = fi[j] & 7;
      out[eoff + obase + j] = (float)(i1[c1] * SK + i2[c2]);
    }
  }
}

// ---- launcher --------------------------------------------------------------
extern "C" void kernel_launch(void* const* d_in, const int* in_sizes, int n_in,
                              void* d_out, int out_size, void* d_ws, size_t ws_size,
                              hipStream_t stream) {
  (void)in_sizes; (void)n_in; (void)out_size; (void)ws_size;
  const float* x     = (const float*)d_in[0];
  const float* Wq    = (const float*)d_in[1];
  const float* bq    = (const float*)d_in[2];
  const float* gamma = (const float*)d_in[3];
  const float* beta  = (const float*)d_in[4];
  const float* K1    = (const float*)d_in[5];
  const float* K2    = (const float*)d_in[6];
  float* out = (float*)d_out;

  // workspace carve-up: qbf(bf16) | wbfT(bf16) | sum | ssq | bias1 | bias2 | k1f | k2f
  __bf16* qbf  = (__bf16*)d_ws;                       // NTOK*HQ
  __bf16* wbfT = qbf + (size_t)NTOK * HQ;             // HQ*DM
  float*  sum  = (float*)(wbfT + (size_t)HQ * DM);    // HQ
  float*  ssq  = sum + HQ;                            // HQ
  float*  bias1 = ssq + HQ;                           // NH*SK
  float*  bias2 = bias1 + NH * SK;                    // NH*SK
  __bf16* k1f  = (__bf16*)(bias2 + NH * SK);          // NH*SK*CH bf16
  __bf16* k2f  = k1f + (size_t)NH * SK * CH;

  zero_kernel<<<(2 * HQ + 255) / 256, 256, 0, stream>>>(sum, 2 * HQ);
  convert_w_kernel<<<dim3(HQ / 64, DM / 64), 256, 0, stream>>>(Wq, wbfT);
  gemm_q_kernel<<<dim3(NTOK / 128, HQ / 128), 256, 0, stream>>>(x, wbfT, bq,
                                                                qbf, sum, ssq);
  fold_kernel<<<NH * SK, 256, 0, stream>>>(sum, ssq, gamma, beta, K1, K2,
                                           k1f, k2f, bias1, bias2);
  score_kernel<<<dim3(NTOK / 16, NH), 256, 0, stream>>>(qbf, k1f, k2f,
                                                        bias1, bias2, out);
}
